// CrossAttentionBlock_41669772705964
// MI455X (gfx1250) — matrix operand order repaired and checked by reading, not compiled, not run on hardware
//
#include <hip/hip_runtime.h>
#include <hip/hip_bf16.h>

typedef _Float16 h16;
typedef __attribute__((ext_vector_type(16))) _Float16 v16h;
typedef __attribute__((ext_vector_type(8)))  _Float16 v8h;
typedef __attribute__((ext_vector_type(8)))  float    v8f;

#define Bz   4
#define LQz  1024
#define LKVz 1344
#define LTz  320
#define Cz   768
#define Hz   12
#define DHz  64
#define HIDz 3072
#define SCALEz 0.125f   // DH^-0.5 = 64^-0.5

// ---------------------------------------------------------------------------
// WMMA fragment helpers (layouts per CDNA5 ISA 7.12.2, wave32)
// A 16x32 f16: lane m = lane&15, half = lane>>4;
//   elems 0..7  -> K = half*8 + e          (VGPR0..3)
//   elems 8..15 -> K = 16 + half*8 + (e-8) (VGPR4..7)
// B 32x16 f16: lane n = lane&15; same K mapping as A
// C/D 16x16 f32: reg r -> (M = r + 8*half, N = lane&15)
// ---------------------------------------------------------------------------
__device__ inline v16h load_a_frag(const h16* __restrict__ A, int lda,
                                   int m_base, int k_base) {
    const int lane = threadIdx.x & 31;
    const int hh   = lane >> 4;
    const int m    = m_base + (lane & 15);
    const h16* p = A + (size_t)m * lda + k_base + hh * 8;
    v8h lo = *(const v8h*)p;
    v8h hi = *(const v8h*)(p + 16);
    v16h r;
#pragma unroll
    for (int i = 0; i < 8; ++i) { r[i] = lo[i]; r[8 + i] = hi[i]; }
    return r;
}

// Bt is the B matrix stored transposed: Bt[n][k], row-major N x K
__device__ inline v16h load_b_frag(const h16* __restrict__ Bt, int ldb,
                                   int n_base, int k_base) {
    const int lane = threadIdx.x & 31;
    const int hh   = lane >> 4;
    const int n    = n_base + (lane & 15);
    const h16* p = Bt + (size_t)n * ldb + k_base + hh * 8;
    v8h lo = *(const v8h*)p;
    v8h hi = *(const v8h*)(p + 16);
    v16h r;
#pragma unroll
    for (int i = 0; i < 8; ++i) { r[i] = lo[i]; r[8 + i] = hi[i]; }
    return r;
}

__device__ inline v8f wmma_f16(v16h a, v16h b, v8f c) {
    return __builtin_amdgcn_wmma_f32_16x16x32_f16(false, a, false, b,
                                                  (short)0, c, false, false);
}

// ---------------------------------------------------------------------------
// Generic WMMA GEMM:  out = A[MxK] @ Bt[NxK]^T + bias, with epilogues.
// block tile 128(M) x 64(N), 8 waves, wave tile 32x32, K step 32.
// ---------------------------------------------------------------------------
enum { EPI_H16_BIAS = 0, EPI_F32_BIAS_RES = 1, EPI_H16_BIAS_GELU = 2 };

template <int EPI>
__global__ __launch_bounds__(256) void gemm_wmma(
    const h16* __restrict__ A, const h16* __restrict__ Bt,
    const float* __restrict__ bias, const float* __restrict__ residual,
    h16* __restrict__ outh, float* __restrict__ outf,
    int M, int N, int K)
{
    const int wave = threadIdx.x >> 5;
    const int lane = threadIdx.x & 31;
    const int hh   = lane >> 4;
    const int l15  = lane & 15;
    const int m0 = blockIdx.y * 128 + (wave >> 1) * 32;
    const int n0 = blockIdx.x * 64  + (wave & 1) * 32;

    v8f acc[2][2];
#pragma unroll
    for (int i = 0; i < 2; ++i)
#pragma unroll
        for (int j = 0; j < 2; ++j) acc[i][j] = (v8f){};

    for (int k = 0; k < K; k += 32) {
        if (k + 64 <= K) {   // pull next k-slab toward the WGP
            __builtin_prefetch(A + (size_t)(m0 + (lane & 15)) * K + k + 32, 0, 1);
            __builtin_prefetch(A + (size_t)(m0 + 16 + (lane & 15)) * K + k + 32, 0, 1);
            __builtin_prefetch(Bt + (size_t)(n0 + (lane & 15)) * K + k + 32, 0, 1);
            __builtin_prefetch(Bt + (size_t)(n0 + 16 + (lane & 15)) * K + k + 32, 0, 1);
        }
        v16h a0 = load_a_frag(A, K, m0,      k);
        v16h a1 = load_a_frag(A, K, m0 + 16, k);
        v16h b0 = load_b_frag(Bt, K, n0,      k);
        v16h b1 = load_b_frag(Bt, K, n0 + 16, k);
        acc[0][0] = wmma_f16(a0, b0, acc[0][0]);
        acc[0][1] = wmma_f16(a0, b1, acc[0][1]);
        acc[1][0] = wmma_f16(a1, b0, acc[1][0]);
        acc[1][1] = wmma_f16(a1, b1, acc[1][1]);
    }

#pragma unroll
    for (int nf = 0; nf < 2; ++nf) {
        const int ng = n0 + nf * 16 + l15;
        const float bz = bias ? bias[ng] : 0.0f;
#pragma unroll
        for (int mf = 0; mf < 2; ++mf) {
#pragma unroll
            for (int r = 0; r < 8; ++r) {
                const int mg = m0 + mf * 16 + r + hh * 8;
                float v = acc[mf][nf][r] + bz;
                const size_t o = (size_t)mg * N + ng;
                if (EPI == EPI_F32_BIAS_RES) {
                    outf[o] = v + residual[o];
                } else if (EPI == EPI_H16_BIAS_GELU) {
                    float g = 0.5f * v * (1.0f + erff(v * 0.70710678118f));
                    outh[o] = (h16)g;
                } else {
                    outh[o] = (h16)v;
                }
            }
        }
    }
}

// ---------------------------------------------------------------------------
// LayerNorm (row length C=768) f32 -> f16, one block per row
// ---------------------------------------------------------------------------
__global__ __launch_bounds__(256) void layernorm_to_f16(
    const float* __restrict__ x, const float* __restrict__ g,
    const float* __restrict__ b, h16* __restrict__ out)
{
    __shared__ float ssum[256];
    __shared__ float ssq[256];
    const int tid = threadIdx.x;
    const float* xr = x + (size_t)blockIdx.x * Cz;
    float a0 = xr[tid], a1 = xr[tid + 256], a2 = xr[tid + 512];
    ssum[tid] = a0 + a1 + a2;
    ssq[tid]  = a0 * a0 + a1 * a1 + a2 * a2;
    __syncthreads();
    for (int off = 128; off > 0; off >>= 1) {
        if (tid < off) { ssum[tid] += ssum[tid + off]; ssq[tid] += ssq[tid + off]; }
        __syncthreads();
    }
    const float mean = ssum[0] * (1.0f / Cz);
    const float var  = ssq[0] * (1.0f / Cz) - mean * mean;
    const float rstd = rsqrtf(var + 1e-5f);
    h16* orow = out + (size_t)blockIdx.x * Cz;
    orow[tid]       = (h16)((a0 - mean) * rstd * g[tid]       + b[tid]);
    orow[tid + 256] = (h16)((a1 - mean) * rstd * g[tid + 256] + b[tid + 256]);
    orow[tid + 512] = (h16)((a2 - mean) * rstd * g[tid + 512] + b[tid + 512]);
}

// ---------------------------------------------------------------------------
// Tiled weight transpose + f32->f16:  in [K][N] -> out [N][K]
// 32x32 tiles via LDS; coalesced on both sides. K,N multiples of 32.
// ---------------------------------------------------------------------------
__global__ __launch_bounds__(256) void transpose_to_f16_tiled(
    const float* __restrict__ in, h16* __restrict__ out, int K, int N)
{
    __shared__ float t[32][33];
    const int tx = threadIdx.x;        // 0..31
    const int ty = threadIdx.y;        // 0..7
    const int n0 = blockIdx.x * 32;
    const int k0 = blockIdx.y * 32;
#pragma unroll
    for (int j = 0; j < 32; j += 8)
        t[ty + j][tx] = in[(size_t)(k0 + ty + j) * N + n0 + tx];
    __syncthreads();
#pragma unroll
    for (int j = 0; j < 32; j += 8)
        out[(size_t)(n0 + ty + j) * K + k0 + tx] = (h16)t[tx][ty + j];
}

// Add tem_mask[b][t][c] (f32) to V block of KVh: KVh[(b*LKV+t)][C + c]
__global__ void add_mask_kernel(const float* __restrict__ tm,
                                h16* __restrict__ KVh)
{
    int idx = blockIdx.x * 256 + threadIdx.x;
    if (idx >= Bz * LTz * Cz) return;
    int c = idx % Cz;
    int t = (idx / Cz) % LTz;
    int b = idx / (LTz * Cz);
    size_t o = ((size_t)(b * LKVz + t)) * (2 * Cz) + Cz + c;
    KVh[o] = (h16)((float)KVh[o] + tm[idx]);
}

// ---------------------------------------------------------------------------
// Flash-style attention. grid = (LQ/128, B*H). Each wave: 16 q-rows.
// kv tile = 64. S = Q Kt * scale + pos ; online softmax ; O += P V.
// ---------------------------------------------------------------------------
__device__ inline float wave16_max(float v) {
#pragma unroll
    for (int m = 1; m <= 8; m <<= 1) v = fmaxf(v, __shfl_xor(v, m, 32));
    return v;
}
__device__ inline float wave16_sum(float v) {
#pragma unroll
    for (int m = 1; m <= 8; m <<= 1) v += __shfl_xor(v, m, 32);
    return v;
}

#define KVSTEP 64

__global__ __launch_bounds__(256) void attention_kernel(
    const h16* __restrict__ Qh, const h16* __restrict__ KVh,
    const float* __restrict__ pos, h16* __restrict__ Xo)
{
    __shared__ __align__(32) h16 Vt[64 * KVSTEP];        // Vt[dh][kv]  8KB
    __shared__ __align__(32) h16 Pl[8 * 16 * KVSTEP];    // per-wave P 16KB

    const int wave = threadIdx.x >> 5;
    const int lane = threadIdx.x & 31;
    const int hh   = lane >> 4;
    const int l15  = lane & 15;
    const int b = blockIdx.y / Hz;
    const int h = blockIdx.y % Hz;
    const int q0 = blockIdx.x * 128 + wave * 16;

    const h16* Qb = Qh + (size_t)b * LQz * Cz;
    const h16* Kb = KVh + (size_t)b * LKVz * (2 * Cz) + h * DHz;
    const h16* Vb = KVh + (size_t)b * LKVz * (2 * Cz) + Cz + h * DHz;
    const float* posb = pos + (size_t)h * LQz * LKVz;

    // Q fragments live in registers for the whole kv loop
    const v16h aq0 = load_a_frag(Qb, Cz, q0, h * DHz);
    const v16h aq1 = load_a_frag(Qb, Cz, q0, h * DHz + 32);

    float row_m[8], row_l[8];
    v8f accO[4];
#pragma unroll
    for (int r = 0; r < 8; ++r) { row_m[r] = -1e30f; row_l[r] = 0.0f; }
#pragma unroll
    for (int f = 0; f < 4; ++f) accO[f] = (v8f){};

    for (int kv0 = 0; kv0 < LKVz; kv0 += KVSTEP) {
        // prefetch next tile's K / V rows and pos cachelines
        if (kv0 + KVSTEP < LKVz) {
            const int nkv = kv0 + KVSTEP;
            __builtin_prefetch(Kb + (size_t)(nkv + lane) * (2 * Cz), 0, 1);
            __builtin_prefetch(Kb + (size_t)(nkv + 32 + lane) * (2 * Cz), 0, 1);
            __builtin_prefetch(Vb + (size_t)(nkv + lane) * (2 * Cz), 0, 1);
            __builtin_prefetch(Vb + (size_t)(nkv + 32 + lane) * (2 * Cz), 0, 1);
            __builtin_prefetch(posb + (size_t)(q0 + l15) * LKVz + nkv + hh * 32, 0, 1);
        }
        // stage V tile transposed into LDS: Vt[dh][kv]
        {
            const int kv_i  = threadIdx.x >> 2;        // 0..63
            const int dh16  = (threadIdx.x & 3) * 16;  // 0,16,32,48
            v16h vv = *(const v16h*)(Vb + (size_t)(kv0 + kv_i) * (2 * Cz) + dh16);
#pragma unroll
            for (int j = 0; j < 16; ++j) Vt[(dh16 + j) * KVSTEP + kv_i] = vv[j];
        }
        __syncthreads();

        // S tiles (16 q x 64 kv)
        v8f s[4];
#pragma unroll
        for (int nf = 0; nf < 4; ++nf) {
            s[nf] = (v8f){};
            s[nf] = wmma_f16(aq0, load_b_frag(Kb, 2 * Cz, kv0 + nf * 16, 0),  s[nf]);
            s[nf] = wmma_f16(aq1, load_b_frag(Kb, 2 * Cz, kv0 + nf * 16, 32), s[nf]);
        }

        // online softmax; exp(S-m) written back into s[]
#pragma unroll
        for (int r = 0; r < 8; ++r) {
            const int qrow = q0 + r + hh * 8;
            const float* pr = posb + (size_t)qrow * LKVz + kv0;
            float v0 = s[0][r] * SCALEz + pr[l15];
            float v1 = s[1][r] * SCALEz + pr[16 + l15];
            float v2 = s[2][r] * SCALEz + pr[32 + l15];
            float v3 = s[3][r] * SCALEz + pr[48 + l15];
            float tm = wave16_max(fmaxf(fmaxf(v0, v1), fmaxf(v2, v3)));
            float mn = fmaxf(row_m[r], tm);
            float corr = __expf(row_m[r] - mn);
            float e0 = __expf(v0 - mn);
            float e1 = __expf(v1 - mn);
            float e2 = __expf(v2 - mn);
            float e3 = __expf(v3 - mn);
            float ts = wave16_sum((e0 + e1) + (e2 + e3));
            row_l[r] = row_l[r] * corr + ts;
            row_m[r] = mn;
#pragma unroll
            for (int f = 0; f < 4; ++f) accO[f][r] *= corr;
            s[0][r] = e0; s[1][r] = e1; s[2][r] = e2; s[3][r] = e3;
        }

        // reshape P into A-fragment layout via LDS (per-wave region)
        h16* Pw = Pl + wave * (16 * KVSTEP);
#pragma unroll
        for (int r = 0; r < 8; ++r) {
            const int mrow = r + hh * 8;
#pragma unroll
            for (int nf = 0; nf < 4; ++nf)
                Pw[mrow * KVSTEP + nf * 16 + l15] = (h16)s[nf][r];
        }
        // LDS ops from the same wave are in-order (DScnt); safe to reload
        const v16h pa0 = load_a_frag(Pw, KVSTEP, 0, 0);
        const v16h pa1 = load_a_frag(Pw, KVSTEP, 0, 32);
#pragma unroll
        for (int f = 0; f < 4; ++f) {
            accO[f] = wmma_f16(pa0, load_b_frag(Vt, KVSTEP, f * 16, 0),  accO[f]);
            accO[f] = wmma_f16(pa1, load_b_frag(Vt, KVSTEP, f * 16, 32), accO[f]);
        }
        __syncthreads();   // protect Vt before next iteration restages it
    }

#pragma unroll
    for (int r = 0; r < 8; ++r) {
        const float inv = 1.0f / row_l[r];
        const int qrow = q0 + r + hh * 8;
        h16* orow = Xo + ((size_t)b * LQz + qrow) * Cz + h * DHz;
#pragma unroll
        for (int f = 0; f < 4; ++f)
            orow[f * 16 + l15] = (h16)(accO[f][r] * inv);
    }
}

// ---------------------------------------------------------------------------
// Host orchestration
// ---------------------------------------------------------------------------
extern "C" void kernel_launch(void* const* d_in, const int* in_sizes, int n_in,
                              void* d_out, int out_size, void* d_ws, size_t ws_size,
                              hipStream_t stream)
{
    const float* tem_mask = (const float*)d_in[0];
    const float* q_in     = (const float*)d_in[1];
    const float* kv_in    = (const float*)d_in[2];
    const float* attn_pos = (const float*)d_in[3];
    const float* ln1q_g   = (const float*)d_in[4];
    const float* ln1q_b   = (const float*)d_in[5];
    const float* ln1kv_g  = (const float*)d_in[6];
    const float* ln1kv_b  = (const float*)d_in[7];
    const float* Wq       = (const float*)d_in[8];
    const float* bq       = (const float*)d_in[9];
    const float* Wkv      = (const float*)d_in[10];
    const float* bkv      = (const float*)d_in[11];
    const float* Wp       = (const float*)d_in[12];
    const float* bp       = (const float*)d_in[13];
    const float* ln2_g    = (const float*)d_in[14];
    const float* ln2_b    = (const float*)d_in[15];
    const float* W1       = (const float*)d_in[16];
    const float* b1       = (const float*)d_in[17];
    const float* W2       = (const float*)d_in[18];
    const float* b2       = (const float*)d_in[19];

    const int MQ = Bz * LQz;    // 4096
    const int MK = Bz * LKVz;   // 5376

    char* ws = (char*)d_ws;
    size_t off = 0;
    auto carve = [&](size_t bytes) -> char* {
        char* p = ws + off;
        off += (bytes + 255) & ~(size_t)255;
        return p;
    };
    h16* WqT  = (h16*)carve((size_t)Cz * Cz * 2);
    h16* WkvT = (h16*)carve((size_t)Cz * 2 * Cz * 2);
    h16* WpT  = (h16*)carve((size_t)Cz * Cz * 2);
    h16* W1T  = (h16*)carve((size_t)Cz * HIDz * 2);
    h16* W2T  = (h16*)carve((size_t)HIDz * Cz * 2);
    h16* qn   = (h16*)carve((size_t)MQ * Cz * 2);
    h16* kvn  = (h16*)carve((size_t)MK * Cz * 2);
    h16* Qh   = (h16*)carve((size_t)MQ * Cz * 2);
    h16* KVh  = (h16*)carve((size_t)MK * 2 * Cz * 2);
    h16* Xat  = (h16*)carve((size_t)MQ * Cz * 2);
    float* q2 = (float*)carve((size_t)MQ * Cz * 4);
    h16* hn   = (h16*)carve((size_t)MQ * Cz * 2);
    h16* ffn1 = (h16*)carve((size_t)MQ * HIDz * 2);

    auto gsz = [](size_t n) { return (unsigned)((n + 255) / 256); };
    dim3 tb(32, 8);

    // weight transposes (f32 [K][N] -> f16 [N][K]), LDS-tiled
    transpose_to_f16_tiled<<<dim3(Cz/32,   Cz/32),   tb, 0, stream>>>(Wq,  WqT,  Cz,   Cz);
    transpose_to_f16_tiled<<<dim3(2*Cz/32, Cz/32),   tb, 0, stream>>>(Wkv, WkvT, Cz,   2*Cz);
    transpose_to_f16_tiled<<<dim3(Cz/32,   Cz/32),   tb, 0, stream>>>(Wp,  WpT,  Cz,   Cz);
    transpose_to_f16_tiled<<<dim3(HIDz/32, Cz/32),   tb, 0, stream>>>(W1,  W1T,  Cz,   HIDz);
    transpose_to_f16_tiled<<<dim3(Cz/32,   HIDz/32), tb, 0, stream>>>(W2,  W2T,  HIDz, Cz);

    // LN1
    layernorm_to_f16<<<MQ, 256, 0, stream>>>(q_in,  ln1q_g,  ln1q_b,  qn);
    layernorm_to_f16<<<MK, 256, 0, stream>>>(kv_in, ln1kv_g, ln1kv_b, kvn);

    // Q / KV projections
    gemm_wmma<EPI_H16_BIAS><<<dim3(Cz/64, MQ/128), 256, 0, stream>>>(
        qn, WqT, bq, nullptr, Qh, nullptr, MQ, Cz, Cz);
    gemm_wmma<EPI_H16_BIAS><<<dim3(2*Cz/64, MK/128), 256, 0, stream>>>(
        kvn, WkvT, bkv, nullptr, KVh, nullptr, MK, 2*Cz, Cz);

    // v += tem_mask on first LT rows
    add_mask_kernel<<<gsz((size_t)Bz * LTz * Cz), 256, 0, stream>>>(tem_mask, KVh);

    // attention
    attention_kernel<<<dim3(LQz/128, Bz*Hz), 256, 0, stream>>>(Qh, KVh, attn_pos, Xat);

    // out projection + residual (q2 = q + x @ Wp + bp)
    gemm_wmma<EPI_F32_BIAS_RES><<<dim3(Cz/64, MQ/128), 256, 0, stream>>>(
        Xat, WpT, bp, q_in, nullptr, q2, MQ, Cz, Cz);

    // LN2 -> FFN
    layernorm_to_f16<<<MQ, 256, 0, stream>>>(q2, ln2_g, ln2_b, hn);
    gemm_wmma<EPI_H16_BIAS_GELU><<<dim3(HIDz/64, MQ/128), 256, 0, stream>>>(
        hn, W1T, b1, nullptr, ffn1, nullptr, MQ, HIDz, Cz);
    gemm_wmma<EPI_F32_BIAS_RES><<<dim3(Cz/64, MQ/128), 256, 0, stream>>>(
        ffn1, W2T, b2, q2, nullptr, (float*)d_out, MQ, Cz, HIDz);
}
